// Graph3DBias_15616501088396
// MI455X (gfx1250) — matrix operand order repaired and checked by reading, not compiled, not run on hardware
//
#include <hip/hip_runtime.h>
#include <hip/hip_bf16.h>
#include <math.h>

// ---------------------------------------------------------------- constants
constexpr int Gg   = 16;     // graphs
constexpr int Nn   = 256;    // atoms
constexpr int Kk   = 128;    // gaussian kernels
constexpr int Hh   = 32;     // heads
constexpr int Dd   = 768;    // merge dim
constexpr int NAT  = 128;    // atom types
constexpr int LDH  = 136;    // LDS row stride in halves (pad, multiple of 8)

constexpr size_t OFF_DIST  = 0;
constexpr size_t OFF_DELTA = (size_t)Gg * Nn * Nn;             // 1,048,576
constexpr size_t OFF_ATTN  = OFF_DELTA + (size_t)Gg*Nn*Nn*3;   // 4,194,304
constexpr size_t OFF_MERGE = OFF_ATTN  + (size_t)Gg*Hh*Nn*Nn;  // 37,748,736

typedef __bf16 v16bf __attribute__((ext_vector_type(16)));
typedef float  v8f   __attribute__((ext_vector_type(8)));

union FragCvt { uint4 q[2]; v16bf v; };

__device__ inline unsigned short f2bf(float f) {
    unsigned u = __builtin_bit_cast(unsigned, f);
    u += 0x7FFFu + ((u >> 16) & 1u);          // round-to-nearest-even
    return (unsigned short)(u >> 16);
}
__device__ inline float bf2f(unsigned short h) {
    return __builtin_bit_cast(float, ((unsigned)h) << 16);
}

// Branch-free exact-GELU: erf via Abramowitz-Stegun 7.1.26 (|err|<1.5e-7),
// single v_exp_f32 + v_rcp_f32, no EXEC divergence (vs libm erff).
__device__ inline float gelu_exact(float x) {
    float ax = fabsf(x) * 0.70710678f;                 // |x|/sqrt(2)
    float t  = __builtin_amdgcn_rcpf(1.0f + 0.3275911f * ax);
    float p  = t * (0.254829592f +
               t * (-0.284496736f +
               t * (1.421413741f +
               t * (-1.453152027f +
               t * 1.061405429f))));
    float e  = 1.0f - p * __expf(-ax * ax);
    e = copysignf(e, x);
    return 0.5f * x * (1.0f + e);
}

// A-matrix 16x32 bf16 fragment (ISA 7.12.2): lane<16 -> K {kb..kb+7, kb+16..kb+23},
// lane>=16 -> K {kb+8..kb+15, kb+24..kb+31}; row = mbase + (lane&15)
__device__ inline v16bf load_frag_a(const unsigned short* lds, int mbase, int kbase, int lane) {
    int r  = lane & 15;
    int hi = lane >> 4;
    const unsigned short* row = lds + (mbase + r) * LDH;
    FragCvt f;
    f.q[0] = *(const uint4*)(row + kbase + hi * 8);
    f.q[1] = *(const uint4*)(row + kbase + 16 + hi * 8);
    return f.v;
}
// B-matrix 32x16 bf16 fragment: lanes 0-15 hold K {kb..kb+7, kb+16..kb+23}, lanes 16-31 hold {kb+8..kb+15, kb+24..kb+31},
// column = nbase + (lane&15). Source staged transposed: lds[col][k].
__device__ inline v16bf load_frag_b(const unsigned short* lds, int nbase, int kbase, int lane) {
    int c  = lane & 15;
    int hi = lane >> 4;
    const unsigned short* row = lds + (nbase + c) * LDH + kbase + hi * 8;
    FragCvt f;
    f.q[0] = *(const uint4*)(row);
    f.q[1] = *(const uint4*)(row + 16);
    return f.v;
}

__device__ inline v8f wmma_bf16(v16bf a, v16bf b, v8f c) {
    return __builtin_amdgcn_wmma_f32_16x16x32_bf16(false, a, false, b, (short)0, c, false, false);
}

// K=128 tile MAC: issue all 4 A-frag loads first (one s_wait_dscnt covers all),
// then 4 chained WMMAs back-to-back on the XDL pipe.
__device__ inline v8f mac_k128(const unsigned short* aLds, int mbase,
                               const v16bf b[4], int lane) {
    v16bf a0 = load_frag_a(aLds, mbase,  0, lane);
    v16bf a1 = load_frag_a(aLds, mbase, 32, lane);
    v16bf a2 = load_frag_a(aLds, mbase, 64, lane);
    v16bf a3 = load_frag_a(aLds, mbase, 96, lane);
    v8f acc = {0.f,0.f,0.f,0.f,0.f,0.f,0.f,0.f};
    acc = wmma_bf16(a0, b[0], acc);
    acc = wmma_bf16(a1, b[1], acc);
    acc = wmma_bf16(a2, b[2], acc);
    acc = wmma_bf16(a3, b[3], acc);
    return acc;
}

// ================================================================ kernel 1
// dist + normalized delta_pos (pure elementwise, bandwidth bound)
__global__ __launch_bounds__(256) void geom_kernel(const float* __restrict__ pos,
                                                   const int*   __restrict__ x,
                                                   float* __restrict__ dist_out,
                                                   float* __restrict__ delta_out) {
    int idx = blockIdx.x * 256 + threadIdx.x;          // (g,i,j) flat
    int j = idx & 255, i = (idx >> 8) & 255, g = idx >> 16;
    int ai = x[g * Nn + i], aj = x[g * Nn + j];
    bool pi_pad = (ai == 0), pj_pad = (aj == 0);
    const float* pp = pos + (size_t)g * Nn * 3;
    float pix = pi_pad ? 0.f : pp[i*3+0], piy = pi_pad ? 0.f : pp[i*3+1], piz = pi_pad ? 0.f : pp[i*3+2];
    float pjx = pj_pad ? 0.f : pp[j*3+0], pjy = pj_pad ? 0.f : pp[j*3+1], pjz = pj_pad ? 0.f : pp[j*3+2];
    float dx = pjx - pix, dy = pjy - piy, dz = pjz - piz;
    float sq = dx*dx + dy*dy + dz*dz;
    float dist = sq > 0.f ? sqrtf(sq) : 0.f;
    float inv  = 1.0f / (dist + 1e-5f);
    dist_out[idx] = dist;
    delta_out[(size_t)idx*3 + 0] = dx * inv;
    delta_out[(size_t)idx*3 + 1] = dy * inv;
    delta_out[(size_t)idx*3 + 2] = dz * inv;
}

// ================================================================ kernel 2
// One workgroup per (g, i): edge_feature tile in LDS -> GEMM1 (bf16 WMMA) ->
// GELU -> GEMM2 -> attn_bias (+ pad mask), plus masked column-sum -> sum_ef ws.
__global__ __launch_bounds__(256) void fused_bias_kernel(
        const float* __restrict__ pos,   const int* __restrict__ x,
        const int*   __restrict__ t,     const float* __restrict__ means,
        const float* __restrict__ stds,  const float* __restrict__ mul_w,
        const float* __restrict__ bias_w,const float* __restrict__ w1,
        const float* __restrict__ b1,    const float* __restrict__ w2,
        const float* __restrict__ b2,    float* __restrict__ attn,
        float* __restrict__ sum_ef_ws) {
    __shared__ __align__(16) unsigned short efS [128 * LDH];   // 34.8 KB (128 edges x K)
    __shared__ __align__(16) unsigned short hS  [128 * LDH];   // 34.8 KB
    __shared__ __align__(16) unsigned short w1tS[Kk  * LDH];   // 34.8 KB  w1^T [n][k]
    __shared__ __align__(16) unsigned short w2tS[Hh  * LDH];   // 8.7 KB   w2^T [n][k]
    __shared__ float meanS[Kk], rstdS[Kk], b1S[Kk], b2S[Hh];
    __shared__ int   padS[Nn];

    const int tid  = threadIdx.x;
    const int wv   = tid >> 5, lane = tid & 31;
    const int g    = blockIdx.x >> 8;
    const int i    = blockIdx.x & 255;

    // ---- stage weights / per-graph gaussian params -----------------------
    int tt = t[g];
    if (tid < Kk) {
        meanS[tid] = means[tt * Kk + tid];
        rstdS[tid] = 1.0f / (fabsf(stds[tt * Kk + tid]) + 1e-5f);
        b1S[tid]   = b1[tid];
    }
    if (tid < Hh) b2S[tid] = b2[tid];
    for (int e = tid; e < Kk * Kk; e += 256) {           // w1 (K,K) -> w1^T
        int n = e & 127, k = e >> 7;
        w1tS[n * LDH + k] = f2bf(w1[k * Kk + n]);
    }
    for (int e = tid; e < Kk * Hh; e += 256) {           // w2 (K,H) -> w2^T
        int n = e & 31, k = e >> 5;
        w2tS[n * LDH + k] = f2bf(w2[k * Hh + n]);
    }
    int aj_self = x[g * Nn + tid];
    padS[tid]   = (aj_self == 0);
    const int  ai    = x[g * Nn + i];
    const bool pad_i = (ai == 0);
    const float* pp  = pos + (size_t)g * Nn * 3;
    const float pix = pad_i ? 0.f : pp[i*3+0];
    const float piy = pad_i ? 0.f : pp[i*3+1];
    const float piz = pad_i ? 0.f : pp[i*3+2];
    __syncthreads();

    // hoist the two B-operand register sets (persist across chunks / mt loops)
    v16bf bw1[4], bw2a[4], bw2b[4];
    #pragma unroll
    for (int kk = 0; kk < 4; ++kk) {
        bw1[kk]  = load_frag_b(w1tS, wv * 16, kk * 32, lane);  // GEMM1: nt = wv
        bw2a[kk] = load_frag_b(w2tS, 0,       kk * 32, lane);  // GEMM2: nt = 0
        bw2b[kk] = load_frag_b(w2tS, 16,      kk * 32, lane);  // GEMM2: nt = 1
    }

    float sacc = 0.0f;                     // per-k masked column sum (tid<K)
    const float INV_A = 0.39894256f;       // 1/sqrt(2*3.14159)

    for (int c = 0; c < 2; ++c) {          // edge chunks of 128 (j dimension)
        const int j0 = c * 128;
        // ---- phase 1: edge_feature tile (2 threads per edge row) ---------
        {
            int jl = tid >> 1;
            int j  = j0 + jl;
            int kh = (tid & 1) * 64;
            bool pad_j = padS[j];
            float pjx = pad_j ? 0.f : pp[j*3+0];
            float pjy = pad_j ? 0.f : pp[j*3+1];
            float pjz = pad_j ? 0.f : pp[j*3+2];
            float dx = pjx - pix, dy = pjy - piy, dz = pjz - piz;
            float sq = dx*dx + dy*dy + dz*dz;
            float dist = sq > 0.f ? sqrtf(sq) : 0.f;
            int   et = ai * NAT + x[g * Nn + j];
            float xg = __ldg(&mul_w[et]) * dist + __ldg(&bias_w[et]);
            unsigned short* row = efS + jl * LDH;
            #pragma unroll 8
            for (int k = kh; k < kh + 64; ++k) {
                float rs = rstdS[k];
                float z  = (xg - meanS[k]) * rs;
                row[k]   = f2bf(__expf(-0.5f * z * z) * (INV_A * rs));
            }
        }
        __syncthreads();
        // ---- masked column sum for merge ---------------------------------
        if (tid < Kk) {
            for (int jl = 0; jl < 128; ++jl)
                if (!padS[j0 + jl]) sacc += bf2f(efS[jl * LDH + tid]);
        }
        // ---- GEMM1: ef(128xK) @ w1(KxK) -> gelu -> hS --------------------
        {
            const int nt = wv;                       // 8 waves x 8 col-tiles
            for (int mt = 0; mt < 8; ++mt) {
                v8f acc = mac_k128(efS, mt * 16, bw1, lane);
                int col   = nt * 16 + (lane & 15);
                int rbase = mt * 16 + ((lane >> 4) << 3);
                float bia = b1S[col];
                #pragma unroll
                for (int r = 0; r < 8; ++r)
                    hS[(rbase + r) * LDH + col] = f2bf(gelu_exact(acc[r] + bia));
            }
        }
        __syncthreads();
        // ---- GEMM2: h(128xK) @ w2(KxH) -> attn_bias ----------------------
        for (int q = 0; q < 2; ++q) {
            int tile = wv * 2 + q;                   // 16 tiles (8mt x 2nt)
            int mt = tile >> 1, nt = tile & 1;
            v8f acc = mac_k128(hS, mt * 16, nt ? bw2b : bw2a, lane);
            int hcol  = nt * 16 + (lane & 15);
            int jbase = j0 + mt * 16 + ((lane >> 4) << 3);
            float bb  = b2S[hcol];
            size_t ob = ((((size_t)g * Hh + hcol) * Nn + i) * Nn) + jbase;
            float o[8];
            #pragma unroll
            for (int r = 0; r < 8; ++r)
                o[r] = padS[jbase + r] ? -INFINITY : (acc[r] + bb);
            *(float4*)(attn + ob)     = make_float4(o[0], o[1], o[2], o[3]);
            *(float4*)(attn + ob + 4) = make_float4(o[4], o[5], o[6], o[7]);
        }
        __syncthreads();   // efS/hS reused next chunk
    }
    if (tid < Kk) sum_ef_ws[((size_t)g * Nn + i) * Kk + tid] = sacc;
}

// ================================================================ kernel 3
// merge = sum_ef (4096 x 128) @ we (128 x 768) + be, bf16 WMMA
__global__ __launch_bounds__(256) void merge_kernel(const float* __restrict__ sum_ef,
                                                    const float* __restrict__ we,
                                                    const float* __restrict__ be,
                                                    float* __restrict__ merge) {
    __shared__ __align__(16) unsigned short aT [16  * LDH];    // 4.3 KB
    __shared__ __align__(16) unsigned short wet[128 * LDH];    // 34.8 KB we^T slice
    __shared__ float beS[128];

    const int tid = threadIdx.x, wv = tid >> 5, lane = tid & 31;
    const int mt  = blockIdx.x;          // 256 row tiles of 16
    const int cb  = blockIdx.y;          // 6 column blocks of 128

    for (int e = tid; e < 16 * Kk; e += 256) {
        int r = e >> 7, k = e & 127;
        aT[r * LDH + k] = f2bf(sum_ef[((size_t)mt * 16 + r) * Kk + k]);
    }
    for (int e = tid; e < Kk * 128; e += 256) {
        int d = e & 127, k = e >> 7;
        wet[d * LDH + k] = f2bf(we[(size_t)k * Dd + cb * 128 + d]);
    }
    if (tid < 128) beS[tid] = be[cb * 128 + tid];
    __syncthreads();

    const int nt = wv;                   // 8 waves x 8 col tiles
    v16bf b[4];
    #pragma unroll
    for (int kk = 0; kk < 4; ++kk)
        b[kk] = load_frag_b(wet, nt * 16, kk * 32, lane);
    v8f acc = mac_k128(aT, 0, b, lane);

    int dl = nt * 16 + (lane & 15);
    int d  = cb * 128 + dl;
    int rb = (lane >> 4) << 3;
    #pragma unroll
    for (int r = 0; r < 8; ++r) {
        int gi = mt * 16 + rb + r;
        merge[(size_t)gi * Dd + d] = acc[r] + beS[dl];
    }
}

// ================================================================ launcher
extern "C" void kernel_launch(void* const* d_in, const int* in_sizes, int n_in,
                              void* d_out, int out_size, void* d_ws, size_t ws_size,
                              hipStream_t stream) {
    const float* pos    = (const float*)d_in[0];
    const int*   x      = (const int*)  d_in[1];
    const int*   t      = (const int*)  d_in[2];
    const float* means  = (const float*)d_in[3];
    const float* stds   = (const float*)d_in[4];
    const float* mul_w  = (const float*)d_in[5];
    const float* bias_w = (const float*)d_in[6];
    const float* w1     = (const float*)d_in[7];
    const float* b1     = (const float*)d_in[8];
    const float* w2     = (const float*)d_in[9];
    const float* b2     = (const float*)d_in[10];
    const float* we     = (const float*)d_in[11];
    const float* be     = (const float*)d_in[12];

    float* out      = (float*)d_out;
    float* dist_o   = out + OFF_DIST;
    float* delta_o  = out + OFF_DELTA;
    float* attn_o   = out + OFF_ATTN;
    float* merge_o  = out + OFF_MERGE;
    float* sum_ef   = (float*)d_ws;      // G*N*K floats = 2 MB scratch

    geom_kernel<<<Gg * Nn * Nn / 256, 256, 0, stream>>>(pos, x, dist_o, delta_o);
    fused_bias_kernel<<<Gg * Nn, 256, 0, stream>>>(pos, x, t, means, stds, mul_w,
                                                   bias_w, w1, b1, w2, b2,
                                                   attn_o, sum_ef);
    merge_kernel<<<dim3(Gg * Nn / 16, Dd / 128), 256, 0, stream>>>(sum_ef, we, be, merge_o);
}
